// LinearAttention_71914932404460
// MI455X (gfx1250) — matrix operand order repaired and checked by reading, not compiled, not run on hardware
//
#include <hip/hip_runtime.h>

// Linear attention (elu(x)+1 feature map) for [n=4, l=s=4096, h=16, d=m=64] f32.
// Pipeline: zero_ws -> la_kv (KV & Ksum via f16 WMMA, f32 atomics, double-buffered
// LDS staging) -> la_out (Q@KV^T * Z via f16 WMMA).

typedef __attribute__((ext_vector_type(16))) _Float16 v16h;
typedef __attribute__((ext_vector_type(8)))  _Float16 v8h;
typedef __attribute__((ext_vector_type(2)))  _Float16 v2h;
typedef __attribute__((ext_vector_type(8)))  float    v8f;

#define SEQ    4096
#define HEADS  16
#define DIM    64
#define NH     64
#define CHUNKS 8
#define CHUNK_S (SEQ / CHUNKS)     // 512
#define KSTEP  32                  // WMMA K per step
#define NSTEP  (CHUNK_S / KSTEP)   // 16
#define LDS1   40                  // padded row stride (f16), 80B = 16B multiple
#define LDS2   72                  // padded row stride (f16), 144B = 16B multiple
#define LA_EPS 1e-6f

__device__ __forceinline__ float feature_map(float x) {
    // elu(x)+1 == exp(min(x,0)) + max(x,0): branchless, no VCC chain,
    // exp is a TRANS op that co-executes with the VALU stream.
    return __expf(fminf(x, 0.0f)) + fmaxf(x, 0.0f);
}

// A fragment (16x32 f16, MxK), lane L: M = L%16, g = L/16,
// element e: K = g*8 + (e&7) + (e>=8)*16  -> two contiguous 16B runs.
__device__ __forceinline__ v16h load_frag_a(const _Float16* base, int m, int lda, int lane) {
    const int g = lane >> 4;
    const _Float16* row = base + m * lda;
    v16h a;
    *((v8h*)&a)       = *((const v8h*)(row +      g * 8));
    *(((v8h*)&a) + 1) = *((const v8h*)(row + 16 + g * 8));
    return a;
}

// B fragment (32x16 f16, KxN), lane L: N = L%16, g = L/16,
// element e: K = g*8 + (e&7) + (e>=8)*16  -> two contiguous 16B runs.
__device__ __forceinline__ v16h load_frag_b(const _Float16* base, int nn, int ldb, int lane) {
    const int g = lane >> 4;
    const _Float16* row = base + nn * ldb + g * 8;
    v16h b;
    *((v8h*)&b)       = *((const v8h*)(row));
    *(((v8h*)&b) + 1) = *((const v8h*)(row + 16));
    return b;
}

__global__ void zero_ws_kernel(float4* __restrict__ p, int nq) {
    int i = blockIdx.x * blockDim.x + threadIdx.x;
    if (i < nq) p[i] = (float4){0.0f, 0.0f, 0.0f, 0.0f};
}

// ---------------------------------------------------------------------------
// Kernel 1: KV[m,d] = sum_s phi(K)[s,d] * V[s,m],  Ksum[d] = sum_s phi(K)[s,d]
// grid: (NH, CHUNKS), block: 256 (8 waves). Double-buffered LDS staging,
// float4 global streams, register-carried Ksum, atomic f32 chunk combine.
// ---------------------------------------------------------------------------
__global__ __launch_bounds__(256)
void la_kv_kernel(const float* __restrict__ Kin, const float* __restrict__ Vin,
                  float* __restrict__ kv_ws, float* __restrict__ ksum_ws) {
    __shared__ __attribute__((aligned(32))) _Float16 sK[2][DIM][LDS1]; // [d][s]: B (N=d, K=s)
    __shared__ __attribute__((aligned(32))) _Float16 sV[2][DIM][LDS1]; // [m][s]: A (M=m, K=s)

    const int nh    = blockIdx.x;             // 0..63
    const int chunk = blockIdx.y;             // 0..7
    const int n     = nh >> 4;
    const int h     = nh & 15;
    const int tid   = threadIdx.x;
    const int lane  = tid & 31;
    const int wave  = tid >> 5;

    // tile map: wave owns one m-tile (shared A) and two adjacent d-tiles
    const int tm  = wave >> 1;                // 0..3
    const int td0 = (wave & 1) * 2;           // 0 or 2
    const int td1 = td0 + 1;
    v8f c0 = {}, c1 = {};

    const long base    = ((long)n * SEQ * HEADS + (long)h) * DIM;  // (n, s=0, h, 0)
    const long sStride = (long)HEADS * DIM;

    // staging ownership: 4 d-values x 2 consecutive s-values per thread
    const int d4 = (tid & 15) * 4;            // 0..60
    const int sl = (tid >> 4) * 2;            // 0..30 (even)
    const int chunkBase = chunk * CHUNK_S;

    float4 k0, k1, v0, v1;                    // software-pipeline registers
    float4 ksAcc = {0.0f, 0.0f, 0.0f, 0.0f};  // f32 Ksum partial for d4..d4+3

    auto ldg = [&](int s0) {
        const long g0 = base + (long)(s0 + sl) * sStride + d4;
        k0 = *(const float4*)(Kin + g0);
        k1 = *(const float4*)(Kin + g0 + sStride);
        v0 = *(const float4*)(Vin + g0);
        v1 = *(const float4*)(Vin + g0 + sStride);
    };
    auto sts = [&](int buf) {
        const float f0x = feature_map(k0.x), f1x = feature_map(k1.x);
        const float f0y = feature_map(k0.y), f1y = feature_map(k1.y);
        const float f0z = feature_map(k0.z), f1z = feature_map(k1.z);
        const float f0w = feature_map(k0.w), f1w = feature_map(k1.w);
        ksAcc.x += f0x + f1x;
        ksAcc.y += f0y + f1y;
        ksAcc.z += f0z + f1z;
        ksAcc.w += f0w + f1w;
        *(v2h*)&sK[buf][d4 + 0][sl] = (v2h){(_Float16)f0x, (_Float16)f1x};
        *(v2h*)&sK[buf][d4 + 1][sl] = (v2h){(_Float16)f0y, (_Float16)f1y};
        *(v2h*)&sK[buf][d4 + 2][sl] = (v2h){(_Float16)f0z, (_Float16)f1z};
        *(v2h*)&sK[buf][d4 + 3][sl] = (v2h){(_Float16)f0w, (_Float16)f1w};
        *(v2h*)&sV[buf][d4 + 0][sl] = (v2h){(_Float16)v0.x, (_Float16)v1.x};
        *(v2h*)&sV[buf][d4 + 1][sl] = (v2h){(_Float16)v0.y, (_Float16)v1.y};
        *(v2h*)&sV[buf][d4 + 2][sl] = (v2h){(_Float16)v0.z, (_Float16)v1.z};
        *(v2h*)&sV[buf][d4 + 3][sl] = (v2h){(_Float16)v0.w, (_Float16)v1.w};
    };

    ldg(chunkBase);
    int cur = 0;
    for (int step = 0; step < NSTEP; ++step) {
        sts(cur);                 // this step's tile -> LDS
        __syncthreads();          // staged tile visible to all waves
        if (step + 1 < NSTEP) ldg(chunkBase + (step + 1) * KSTEP);  // overlap next loads

        v16h a  = load_frag_a(&sV[cur][0][0], tm  * 16 + (lane & 15), LDS1, lane);
        v16h b0 = load_frag_b(&sK[cur][0][0], td0 * 16 + (lane & 15), LDS1, lane);
        v16h b1 = load_frag_b(&sK[cur][0][0], td1 * 16 + (lane & 15), LDS1, lane);
        c0 = __builtin_amdgcn_wmma_f32_16x16x32_f16(false, a, false, b0, (short)0, c0, false, false);
        c1 = __builtin_amdgcn_wmma_f32_16x16x32_f16(false, a, false, b1, (short)0, c1, false, false);
        cur ^= 1;
    }

    // combine chunk partials of KV
    float* kv = kv_ws + nh * (DIM * DIM);     // layout [m][d]
    const int g  = lane >> 4;
    const int nn = lane & 15;
#pragma unroll
    for (int r = 0; r < 8; ++r) {
        const int m = tm * 16 + r + 8 * g;
        atomicAdd(&kv[m * DIM + td0 * 16 + nn], c0[r]);
        atomicAdd(&kv[m * DIM + td1 * 16 + nn], c1[r]);
    }

    // Ksum: 16 (s-group) x 64 (d) LDS reduction, then 64 atomics per WG
    __syncthreads();                          // all fragment reads of sK done
    float* red = (float*)&sK[0][0][0];        // 4 KB scratch
    *(float4*)&red[(tid >> 4) * DIM + d4] = ksAcc;
    __syncthreads();
    if (tid < DIM) {
        float s = 0.0f;
#pragma unroll
        for (int i = 0; i < 16; ++i) s += red[i * DIM + tid];
        atomicAdd(&ksum_ws[nh * DIM + tid], s);
    }
}

// ---------------------------------------------------------------------------
// Kernel 2: out[l,m] = (phi(Q)[l,:] . KV[m,:]) * Z[l],  Z = 1/(phi(Q).Ksum + eps)
// grid: n*h*(SEQ/64) = 4096, block: 256 (8 waves), 64 l-rows per workgroup.
// ---------------------------------------------------------------------------
__global__ __launch_bounds__(256)
void la_out_kernel(const float* __restrict__ Qin, const float* __restrict__ kv_ws,
                   const float* __restrict__ ksum_ws, float* __restrict__ Out) {
    __shared__ __attribute__((aligned(32))) _Float16 sA[64][LDS2];  // phi(Q) [l][d]
    __shared__ __attribute__((aligned(32))) _Float16 sB[DIM][LDS2]; // KV [m][d]
    __shared__ float sKs[DIM];
    __shared__ float sZ[64];

    const int bl   = blockIdx.x;
    const int lt   = bl & 63;
    const int h    = (bl >> 6) & 15;
    const int n    = bl >> 10;
    const int nh   = (n << 4) | h;
    const int tid  = threadIdx.x;
    const int lane = tid & 31;
    const int wave = tid >> 5;
    const int l0   = lt * 64;

    const long qbase   = ((long)n * SEQ * HEADS + (long)h) * DIM;
    const long sStride = (long)HEADS * DIM;
    const float* kv = kv_ws + nh * (DIM * DIM);

    // stage KV and phi(Q) as f16, 8 contiguous d per thread (b128 in, b128 out)
#pragma unroll
    for (int i = 0; i < 2; ++i) {
        const int ch = tid + i * 256;         // 0..511
        const int m  = ch >> 3;
        const int d8 = (ch & 7) * 8;
        float4 a0 = *(const float4*)(kv + m * DIM + d8);
        float4 a1 = *(const float4*)(kv + m * DIM + d8 + 4);
        *(v8h*)&sB[m][d8] = (v8h){(_Float16)a0.x, (_Float16)a0.y, (_Float16)a0.z, (_Float16)a0.w,
                                  (_Float16)a1.x, (_Float16)a1.y, (_Float16)a1.z, (_Float16)a1.w};
        const float* qp = Qin + qbase + (long)(l0 + m) * sStride + d8;
        float4 q0 = *(const float4*)(qp);
        float4 q1 = *(const float4*)(qp + 4);
        *(v8h*)&sA[m][d8] = (v8h){(_Float16)feature_map(q0.x), (_Float16)feature_map(q0.y),
                                  (_Float16)feature_map(q0.z), (_Float16)feature_map(q0.w),
                                  (_Float16)feature_map(q1.x), (_Float16)feature_map(q1.y),
                                  (_Float16)feature_map(q1.z), (_Float16)feature_map(q1.w)};
    }
    if (tid < DIM) sKs[tid] = ksum_ws[nh * DIM + tid];
    __syncthreads();

    // normalizer per l-row
    if (tid < 64) {
        float acc = 0.0f;
#pragma unroll
        for (int d = 0; d < DIM; ++d) acc += (float)sA[tid][d] * sKs[d];
        sZ[tid] = 1.0f / (acc + LA_EPS);
    }
    __syncthreads();

    // wave owns one l-tile (shared A) and two adjacent m-tiles
    const int tl  = wave >> 1;
    const int tm0 = (wave & 1) * 2;
    const int tm1 = tm0 + 1;
    v8f c0 = {}, c1 = {};
#pragma unroll
    for (int d0 = 0; d0 < DIM; d0 += KSTEP) {
        v16h a  = load_frag_a(&sA[0][0] + d0, tl  * 16 + (lane & 15), LDS2, lane);
        v16h b0 = load_frag_b(&sB[0][0] + d0, tm0 * 16 + (lane & 15), LDS2, lane);
        v16h b1 = load_frag_b(&sB[0][0] + d0, tm1 * 16 + (lane & 15), LDS2, lane);
        c0 = __builtin_amdgcn_wmma_f32_16x16x32_f16(false, a, false, b0, (short)0, c0, false, false);
        c1 = __builtin_amdgcn_wmma_f32_16x16x32_f16(false, a, false, b1, (short)0, c1, false, false);
    }

    // scale rows by Z and store: out[((n*SEQ + l)*HEADS + h)*DIM + m]
    const int g  = lane >> 4;
    const int nn = lane & 15;
    const long obase = (long)n * SEQ * HEADS * DIM + (long)h * DIM;
#pragma unroll
    for (int r = 0; r < 8; ++r) {
        const int  lr = tl * 16 + r + 8 * g;
        const float z = sZ[lr];
        const long ro = obase + (long)(l0 + lr) * sStride;
        Out[ro + tm0 * 16 + nn] = c0[r] * z;
        Out[ro + tm1 * 16 + nn] = c1[r] * z;
    }
}

extern "C" void kernel_launch(void* const* d_in, const int* in_sizes, int n_in,
                              void* d_out, int out_size, void* d_ws, size_t ws_size,
                              hipStream_t stream) {
    const float* Q = (const float*)d_in[0];
    const float* K = (const float*)d_in[1];
    const float* V = (const float*)d_in[2];
    float* Out = (float*)d_out;

    float* kv_ws   = (float*)d_ws;               // NH*64*64 floats
    float* ksum_ws = kv_ws + NH * DIM * DIM;     // NH*64 floats
    const int nz = NH * DIM * DIM + NH * DIM;    // 266,240 floats (div by 4)

    zero_ws_kernel<<<(nz / 4 + 255) / 256, 256, 0, stream>>>((float4*)kv_ws, nz / 4);

    dim3 g1(NH, CHUNKS);
    la_kv_kernel<<<g1, 256, 0, stream>>>(K, V, kv_ws, ksum_ws);

    la_out_kernel<<<4096, 256, 0, stream>>>(Q, kv_ws, ksum_ws, Out);
}
